// SimpleMambaBlock_51737176048194
// MI455X (gfx1250) — matrix operand order repaired and checked by reading, not compiled, not run on hardware
//
#include <hip/hip_runtime.h>
#include <hip/hip_bf16.h>
#include <cstdint>

#define BATCH   4
#define SEQ     2048
#define DMODEL  1024
#define DINNER  2048
#define ETOT    4096          /* 2*DINNER */
#define BL      (BATCH*SEQ)   /* 8192 rows */
#define DTRANK  64
#define NBC     96            /* DTRANK + 2*NSTATE */
#define NSTATE  16
#define DCONV   4

typedef __attribute__((ext_vector_type(16))) __bf16       v16bf;
typedef __attribute__((ext_vector_type(8)))  float        v8f;
typedef __attribute__((ext_vector_type(4)))  unsigned int u32x4;

union FragU { v16bf v; u32x4 q[2]; };

__device__ __forceinline__ unsigned short f32_to_bf16(float f) {
  unsigned int u = __float_as_uint(f);
  u += 0x7fffu + ((u >> 16) & 1u);           // round to nearest even
  return (unsigned short)(u >> 16);
}
__device__ __forceinline__ float bf16_to_f32(unsigned short u) {
  return __uint_as_float(((unsigned int)u) << 16);
}
__device__ __forceinline__ float sigmoid_(float v) { return 1.0f / (1.0f + __expf(-v)); }

// ---------------------------------------------------------------------------
// CDNA5 async global->LDS copy (ASYNCcnt path, GV mode).
// LDS byte offset = low 32 bits of the flat shared pointer.
// ---------------------------------------------------------------------------
__device__ __forceinline__ void async_ld_b128(void* lds_dst, const void* gsrc) {
  unsigned l = (unsigned)(uintptr_t)lds_dst;
  asm volatile("global_load_async_to_lds_b128 %0, %1, off"
               :: "v"(l), "v"(gsrc) : "memory");
}
__device__ __forceinline__ void wait_async0() {
  asm volatile("s_wait_asynccnt 0x0" ::: "memory");
}

// ---------------------------------------------------------------------------
// WMMA fragment loaders (gfx1250 wave32, 16-bit element layouts per CDNA5 ISA)
// A: 16x32 (MxK).  lane<16: row=lane,    K chunks {0..7},{16..23}
//                  lane>=16: row=lane-16, K chunks {8..15},{24..31}
// ---------------------------------------------------------------------------
__device__ __forceinline__ v16bf load_fragA(const unsigned short* p0, int ld, int lane) {
  int r  = lane & 15;
  int kh = (lane & 16) ? 8 : 0;
  const unsigned short* p = p0 + (size_t)r * ld + kh;
  FragU f;
  f.q[0] = *(const u32x4*)(p);
  f.q[1] = *(const u32x4*)(p + 16);
  return f.v;
}
// B: 32x16 (KxN), built from W stored row-major (N,K):
//   lane<16: col n=lane,    K chunks {0..7},{16..23} ; lane>=16: col n=lane-16, K chunks {8..15},{24..31}
__device__ __forceinline__ v16bf load_fragB(const unsigned short* p0, int ld, int lane) {
  int n  = lane & 15;
  int kh = (lane & 16) ? 8 : 0;
  const unsigned short* p = p0 + (size_t)n * ld + kh;
  FragU f;
  f.q[0] = *(const u32x4*)(p);
  f.q[1] = *(const u32x4*)(p + 16);
  return f.v;
}

// ---------------------------------------------------------------------------
// Generic bf16 WMMA GEMM:  C[M,N] = A[M,K] * W[N,K]^T   (+ optional residual)
// MODE 0: store f32   MODE 1: store bf16   MODE 2: store f32 + res
// One wave computes a 32 x (16*WTN) C tile (2 x WTN grid of 16x16 WMMAs).
// ---------------------------------------------------------------------------
template <int MODE, int WTN>
__global__ __launch_bounds__(256) void gemm_bf16_kernel(
    const unsigned short* __restrict__ A, const unsigned short* __restrict__ W,
    float* __restrict__ Cf, unsigned short* __restrict__ Cb,
    const float* __restrict__ res, int M, int N, int K) {
  const int TN = 16 * WTN;
  int lane = threadIdx.x & 31;
  int wave = threadIdx.x >> 5;
  int tile = blockIdx.x * 8 + wave;
  int tn_cnt = N / TN;
  int total  = (M >> 5) * tn_cnt;
  if (tile >= total) return;
  int tm = tile / tn_cnt;
  int tn = tile - tm * tn_cnt;
  int m0 = tm << 5, n0 = tn * TN;

  const unsigned short* Ab = A + (size_t)m0 * K;
  const unsigned short* Wb = W + (size_t)n0 * K;

  v8f acc[2][WTN] = {};

  for (int k0 = 0; k0 < K; k0 += 32) {
    __builtin_prefetch(Ab + k0 + 64, 0, 0);
    __builtin_prefetch(Wb + k0 + 64, 0, 0);
    v16bf a0 = load_fragA(Ab + k0, K, lane);
    v16bf a1 = load_fragA(Ab + (size_t)16 * K + k0, K, lane);
    v16bf bfrag[WTN];
#pragma unroll
    for (int j = 0; j < WTN; j++)
      bfrag[j] = load_fragB(Wb + (size_t)(16 * j) * K + k0, K, lane);
#pragma unroll
    for (int j = 0; j < WTN; j++) {
      acc[0][j] = __builtin_amdgcn_wmma_f32_16x16x32_bf16(false, a0, false, bfrag[j],
                                                          (short)0, acc[0][j], false, false);
      acc[1][j] = __builtin_amdgcn_wmma_f32_16x16x32_bf16(false, a1, false, bfrag[j],
                                                          (short)0, acc[1][j], false, false);
    }
  }

  // C/D layout: lane<16 -> rows +0..7 ; lane>=16 -> rows +8..15 ; col = (lane&15)
  int n  = lane & 15;
  int mb = (lane & 16) ? 8 : 0;
#pragma unroll
  for (int i = 0; i < 2; i++) {
#pragma unroll
    for (int j = 0; j < WTN; j++) {
      int mm = m0 + i * 16 + mb;
      int nn = n0 + j * 16 + n;
#pragma unroll
      for (int r = 0; r < 8; r++) {
        size_t idx = (size_t)(mm + r) * N + nn;
        if (MODE == 0) Cf[idx] = acc[i][j][r];
        else if (MODE == 1) Cb[idx] = f32_to_bf16(acc[i][j][r]);
        else Cf[idx] = res[idx] + acc[i][j][r];
      }
    }
  }
}

// ---------------------------------------------------------------------------
// LayerNorm over rows of DMODEL, output bf16
// ---------------------------------------------------------------------------
__global__ __launch_bounds__(256) void ln_kernel(
    const float* __restrict__ x, const float* __restrict__ w,
    const float* __restrict__ b, unsigned short* __restrict__ h) {
  int row = blockIdx.x;
  const float* xr = x + (size_t)row * DMODEL;
  float s = 0.f, s2 = 0.f;
  for (int i = threadIdx.x; i < DMODEL; i += 256) {
    float v = xr[i]; s += v; s2 += v * v;
  }
  __shared__ float red[18];
#pragma unroll
  for (int off = 16; off > 0; off >>= 1) {
    s  += __shfl_down(s,  off);
    s2 += __shfl_down(s2, off);
  }
  int wave = threadIdx.x >> 5, lane = threadIdx.x & 31;
  if (lane == 0) { red[wave] = s; red[8 + wave] = s2; }
  __syncthreads();
  if (threadIdx.x == 0) {
    float a = 0.f, c = 0.f;
    for (int i = 0; i < 8; i++) { a += red[i]; c += red[8 + i]; }
    red[16] = a; red[17] = c;
  }
  __syncthreads();
  float mean = red[16] * (1.0f / DMODEL);
  float var  = red[17] * (1.0f / DMODEL) - mean * mean;
  float rstd = rsqrtf(var + 1e-5f);
  for (int i = threadIdx.x; i < DMODEL; i += 256) {
    float v = (xr[i] - mean) * rstd * w[i] + b[i];
    h[(size_t)row * DMODEL + i] = f32_to_bf16(v);
  }
}

// elementwise f32 -> bf16
__global__ __launch_bounds__(256) void cvt_bf16_kernel(
    const float* __restrict__ in, unsigned short* __restrict__ out, int n) {
  int i = blockIdx.x * 256 + threadIdx.x;
  if (i < n) out[i] = f32_to_bf16(in[i]);
}

// ---------------------------------------------------------------------------
// Depthwise causal conv(4) + SiLU.  dir=0: taps l-3..l (fwd causal)
// dir=1 (reverse branch at forward positions): taps l..l+3, reversed weights
// Input: xz bf16 (BL, ETOT), x part = cols [0, DINNER). Output bf16 (BL, DINNER)
// ---------------------------------------------------------------------------
__global__ __launch_bounds__(256) void conv_silu_kernel(
    const unsigned short* __restrict__ xz, const float* __restrict__ cw,
    const float* __restrict__ cb, unsigned short* __restrict__ xc, int dir) {
  size_t idx = (size_t)blockIdx.x * 256 + threadIdx.x;
  if (idx >= (size_t)BL * DINNER) return;
  int d = (int)(idx % DINNER);
  size_t m = idx / DINNER;
  int l = (int)(m % SEQ);
  size_t brow = m - l;  // first row of this batch
  float acc = cb[d];
#pragma unroll
  for (int j = 0; j < DCONV; j++) {
    int lj = dir ? (l + (DCONV - 1) - j) : (l - (DCONV - 1) + j);
    if (lj >= 0 && lj < SEQ) {
      float v = bf16_to_f32(xz[(brow + lj) * (size_t)ETOT + d]);
      acc += cw[d * DCONV + j] * v;
    }
  }
  acc = acc * sigmoid_(acc);
  xc[m * DINNER + d] = f32_to_bf16(acc);
}

// extract dt (first DTRANK cols of x_dbl) as bf16
__global__ __launch_bounds__(256) void extract_dt_kernel(
    const float* __restrict__ xdbl, unsigned short* __restrict__ dtb) {
  int i = blockIdx.x * 256 + threadIdx.x;
  if (i >= BL * DTRANK) return;
  int m = i / DTRANK, c = i - m * DTRANK;
  dtb[i] = f32_to_bf16(xdbl[(size_t)m * NBC + c]);
}

// softplus(v + bias) in place
__global__ __launch_bounds__(256) void softplus_kernel(
    float* __restrict__ dt, const float* __restrict__ bias) {
  size_t i = (size_t)blockIdx.x * 256 + threadIdx.x;
  if (i >= (size_t)BL * DINNER) return;
  int d = (int)(i % DINNER);
  float v = dt[i] + bias[d];
  dt[i] = (v > 20.f) ? v : log1pf(__expf(v));
}

// ---------------------------------------------------------------------------
// Selective scan.  One thread per (b, d): h[16] in registers, loop over SEQ.
// B_t|C_t (32 contiguous floats per step) staged into LDS via CDNA5 async
// global->LDS b128 copies: one copy per thread per 32-step chunk.
// dir=0: forward, y = result;  dir=1: reverse order, y += result.
// ---------------------------------------------------------------------------
__global__ __launch_bounds__(256) void scan_kernel(
    const float* __restrict__ xdbl, const float* __restrict__ dtf,
    const unsigned short* __restrict__ xc, const float* __restrict__ A_log,
    const float* __restrict__ Dp, float* __restrict__ y, int dir) {
  int b = blockIdx.y;
  int d = blockIdx.x * 256 + threadIdx.x;
  float a[NSTATE], h[NSTATE];
#pragma unroll
  for (int n = 0; n < NSTATE; n++) {
    a[n] = -__expf(A_log[(size_t)d * NSTATE + n]);
    h[n] = 0.f;
  }
  float Dd = Dp[d];
  __shared__ __align__(16) float sBC[32][2 * NSTATE];  // [tt][0..15]=B, [16..31]=C

  for (int c = 0; c < SEQ / 32; c++) {
    __syncthreads();
    {
      // 32 steps x 128 B each = 256 x b128: one async copy per thread
      int tt = threadIdx.x >> 3;
      int c8 = threadIdx.x & 7;
      int tq = dir ? (SEQ - 1 - (c * 32 + tt)) : (c * 32 + tt);
      const float* g = xdbl + ((size_t)b * SEQ + tq) * NBC + DTRANK + c8 * 4;
      async_ld_b128(&sBC[tt][c8 * 4], g);
      wait_async0();
    }
    __syncthreads();
    for (int tt = 0; tt < 32; tt++) {
      int tq = dir ? (SEQ - 1 - (c * 32 + tt)) : (c * 32 + tt);
      size_t mi = (size_t)b * SEQ + tq;
      float dtv = dtf[mi * DINNER + d];
      float ut  = bf16_to_f32(xc[mi * DINNER + d]);
      float du  = dtv * ut;
      float yv  = 0.f;
#pragma unroll
      for (int n = 0; n < NSTATE; n++) {
        h[n] = __expf(dtv * a[n]) * h[n] + du * sBC[tt][n];
        yv += h[n] * sBC[tt][NSTATE + n];
      }
      yv += Dd * ut;
      size_t oi = mi * DINNER + d;
      if (dir) y[oi] += yv; else y[oi] = yv;
    }
  }
}

// ysum_bf16 = bf16( (y_f + y_r) * silu(z) ), z = xz cols [DINNER, 2*DINNER)
__global__ __launch_bounds__(256) void gate_kernel(
    const float* __restrict__ y, const unsigned short* __restrict__ xz,
    unsigned short* __restrict__ ys) {
  size_t i = (size_t)blockIdx.x * 256 + threadIdx.x;
  if (i >= (size_t)BL * DINNER) return;
  int d = (int)(i % DINNER);
  size_t m = i / DINNER;
  float z = bf16_to_f32(xz[m * ETOT + DINNER + d]);
  ys[i] = f32_to_bf16(y[i] * z * sigmoid_(z));
}

// ---------------------------------------------------------------------------
extern "C" void kernel_launch(void* const* d_in, const int* in_sizes, int n_in,
                              void* d_out, int out_size, void* d_ws, size_t ws_size,
                              hipStream_t stream) {
  const float* x        = (const float*)d_in[0];
  const float* norm_w   = (const float*)d_in[1];
  const float* norm_b   = (const float*)d_in[2];
  const float* in_proj  = (const float*)d_in[3];
  const float* conv_w[2]   = {(const float*)d_in[4],  (const float*)d_in[11]};
  const float* conv_b[2]   = {(const float*)d_in[5],  (const float*)d_in[12]};
  const float* xproj_w[2]  = {(const float*)d_in[6],  (const float*)d_in[13]};
  const float* dtproj_w[2] = {(const float*)d_in[7],  (const float*)d_in[14]};
  const float* dtproj_b[2] = {(const float*)d_in[8],  (const float*)d_in[15]};
  const float* A_log[2]    = {(const float*)d_in[9],  (const float*)d_in[16]};
  const float* Dparam[2]   = {(const float*)d_in[10], (const float*)d_in[17]};
  const float* out_proj = (const float*)d_in[18];
  float* out = (float*)d_out;

  char* wsb = (char*)d_ws;
  size_t off = 0;
  auto alloc = [&](size_t bytes) -> char* {
    char* p = wsb + off;
    off += (bytes + 255) & ~(size_t)255;
    return p;
  };
  unsigned short* h_bf    = (unsigned short*)alloc((size_t)BL * DMODEL * 2);
  unsigned short* w_in    = (unsigned short*)alloc((size_t)ETOT * DMODEL * 2);
  unsigned short* w_xp[2] = {(unsigned short*)alloc((size_t)NBC * DINNER * 2),
                             (unsigned short*)alloc((size_t)NBC * DINNER * 2)};
  unsigned short* w_dt[2] = {(unsigned short*)alloc((size_t)DINNER * DTRANK * 2),
                             (unsigned short*)alloc((size_t)DINNER * DTRANK * 2)};
  unsigned short* w_out   = (unsigned short*)alloc((size_t)DMODEL * DINNER * 2);
  unsigned short* xz_bf   = (unsigned short*)alloc((size_t)BL * ETOT * 2);
  unsigned short* xconv   = (unsigned short*)alloc((size_t)BL * DINNER * 2);
  float*          xdbl    = (float*)alloc((size_t)BL * NBC * 4);
  unsigned short* dt_bf   = (unsigned short*)alloc((size_t)BL * DTRANK * 2);
  float*          dtfull  = (float*)alloc((size_t)BL * DINNER * 4);
  float*          ybuf    = (float*)alloc((size_t)BL * DINNER * 4);
  unsigned short* ysum    = (unsigned short*)dtfull;  // alias: dtfull dead after scans

  auto cvt = [&](const float* src, unsigned short* dst, int n) {
    cvt_bf16_kernel<<<(n + 255) / 256, 256, 0, stream>>>(src, dst, n);
  };
  auto gemm = [&](int mode, int wtn, const unsigned short* A, const unsigned short* W,
                  float* Cf, unsigned short* Cb, const float* res, int M, int N, int K) {
    int TN = 16 * wtn;
    int tiles = (M / 32) * (N / TN);
    int blocks = (tiles + 7) / 8;
    if (mode == 0 && wtn == 2)
      gemm_bf16_kernel<0, 2><<<blocks, 256, 0, stream>>>(A, W, Cf, Cb, res, M, N, K);
    else if (mode == 0)
      gemm_bf16_kernel<0, 4><<<blocks, 256, 0, stream>>>(A, W, Cf, Cb, res, M, N, K);
    else if (mode == 1)
      gemm_bf16_kernel<1, 4><<<blocks, 256, 0, stream>>>(A, W, Cf, Cb, res, M, N, K);
    else
      gemm_bf16_kernel<2, 4><<<blocks, 256, 0, stream>>>(A, W, Cf, Cb, res, M, N, K);
  };

  // 1) bf16 weight copies
  cvt(in_proj, w_in, ETOT * DMODEL);
  cvt(xproj_w[0], w_xp[0], NBC * DINNER);
  cvt(xproj_w[1], w_xp[1], NBC * DINNER);
  cvt(dtproj_w[0], w_dt[0], DINNER * DTRANK);
  cvt(dtproj_w[1], w_dt[1], DINNER * DTRANK);
  cvt(out_proj, w_out, DMODEL * DINNER);

  // 2) LayerNorm -> bf16
  ln_kernel<<<BL, 256, 0, stream>>>(x, norm_w, norm_b, h_bf);

  // 3) in_proj GEMM -> xz (bf16, stored (b,l,e))
  gemm(1, 4, h_bf, w_in, nullptr, xz_bf, nullptr, BL, ETOT, DMODEL);

  const int NEL = BL * DINNER;
  // 4) both branches (fwd then reverse), reusing intermediates
  for (int dir = 0; dir < 2; dir++) {
    conv_silu_kernel<<<(NEL + 255) / 256, 256, 0, stream>>>(
        xz_bf, conv_w[dir], conv_b[dir], xconv, dir);
    gemm(0, 2, xconv, w_xp[dir], xdbl, nullptr, nullptr, BL, NBC, DINNER);
    extract_dt_kernel<<<(BL * DTRANK + 255) / 256, 256, 0, stream>>>(xdbl, dt_bf);
    gemm(0, 4, dt_bf, w_dt[dir], dtfull, nullptr, nullptr, BL, DINNER, DTRANK);
    softplus_kernel<<<(NEL + 255) / 256, 256, 0, stream>>>(dtfull, dtproj_b[dir]);
    scan_kernel<<<dim3(DINNER / 256, BATCH), 256, 0, stream>>>(
        xdbl, dtfull, xconv, A_log[dir], Dparam[dir], ybuf, dir);
  }

  // 5) gate with silu(z) -> bf16 (aliases dtfull region)
  gate_kernel<<<(NEL + 255) / 256, 256, 0, stream>>>(ybuf, xz_bf, ysum);

  // 6) out_proj GEMM + residual -> d_out (f32)
  gemm(2, 4, ysum, w_out, out, nullptr, x, BL, DMODEL, DINNER);

  (void)in_sizes; (void)n_in; (void)out_size; (void)ws_size;
}
